// ControlledGCN_8632884265211
// MI455X (gfx1250) — matrix operand order repaired and checked by reading, not compiled, not run on hardware
//
#include <hip/hip_runtime.h>
#include <hip/hip_bf16.h>

#define NN 170000
#define NE 1200000
#define IN_D 128
#define HID_D 128
#define OUT_D 64

typedef __attribute__((ext_vector_type(16))) __bf16 v16bf;
typedef __attribute__((ext_vector_type(4)))  __bf16 v4bf;
typedef __attribute__((ext_vector_type(8)))  float  v8f;
typedef __attribute__((ext_vector_type(4)))  float  v4f;

// ---------------- utility kernels ----------------

__global__ void gcn_zero_f32(float* __restrict__ p, int n) {
  int i = blockIdx.x * 256 + threadIdx.x;
  if (i < n) p[i] = 0.0f;
}

// exact-size vec4 zero (call-site guarantees divisibility)
__global__ void gcn_zero_v4(float* __restrict__ p) {
  long i = (long)(blockIdx.x * 256 + threadIdx.x) * 4;
  v4f z = {0.f, 0.f, 0.f, 0.f};
  *(v4f*)(p + i) = z;
}

__global__ void gcn_degree(const int* __restrict__ src, const int* __restrict__ dst,
                           float* __restrict__ deg_out, float* __restrict__ deg_in) {
  int e = blockIdx.x * 256 + threadIdx.x;
  if (e < NE) {
    atomicAdd(&deg_out[src[e]], 1.0f);
    atomicAdd(&deg_in[dst[e]], 1.0f);
  }
}

__global__ void gcn_norm_finalize(float* __restrict__ d, int n) {
  int i = blockIdx.x * 256 + threadIdx.x;
  if (i < n) d[i] = rsqrtf(fmaxf(d[i], 1.0f));
}

// hs = bf16(feat * norm_out[row]); one thread = 4 consecutive features (same row)
__global__ void gcn_scale_feat(const float* __restrict__ feat,
                               const float* __restrict__ norm_out,
                               __bf16* __restrict__ hs) {
  long i4 = (long)(blockIdx.x * 256 + threadIdx.x) * 4;
  int row = (int)(i4 >> 7);
  float s = norm_out[row];
  v4f f = *(const v4f*)(feat + i4);
  v4bf o;
  o[0] = (__bf16)(f[0] * s);
  o[1] = (__bf16)(f[1] * s);
  o[2] = (__bf16)(f[2] * s);
  o[3] = (__bf16)(f[3] * s);
  *(v4bf*)(hs + i4) = o;
}

// pack W[128 x ncols] f32 into WMMA B-matrix lane layout (bf16):
// Wp[((kt*ntiles + nt)*32 + lane)*16 + i] = bf16( W[kt*32 + (i>>3)*16 + (lane>>4)*8 + (i&7)][nt*16 + (lane&15)] )
__global__ void gcn_pack_w(const float* __restrict__ W, __bf16* __restrict__ Wp, int ncols) {
  int idx = blockIdx.x * 256 + threadIdx.x;
  if (idx >= 128 * ncols) return;
  int i    = idx & 15;
  int lane = (idx >> 4) & 31;
  int tile = idx >> 9;                // kt*ntiles + nt
  int ntiles = ncols >> 4;
  int kt = tile / ntiles;
  int nt = tile - kt * ntiles;
  int k = kt * 32 + (i >> 3) * 16 + (lane >> 4) * 8 + (i & 7);
  int c = nt * 16 + (lane & 15);
  Wp[idx] = (__bf16)(W[k * ncols + c]);
}

// one wave per edge; 4 features per lane; bf16 gather, f32 atomic scatter-add (L2-resident agg)
__global__ void gcn_edge_scatter(const int* __restrict__ src, const int* __restrict__ dst,
                                 const __bf16* __restrict__ hs, float* __restrict__ agg) {
  int e = blockIdx.x * 8 + (threadIdx.x >> 5);
  int lane = threadIdx.x & 31;
  int s = src[e];
  int d = dst[e];
  v4bf v = *(const v4bf*)(hs + (long)s * 128 + lane * 4);
  float* base = agg + (long)d * 128 + lane * 4;
  atomicAdd(base + 0, (float)v[0]);
  atomicAdd(base + 1, (float)v[1]);
  atomicAdd(base + 2, (float)v[2]);
  atomicAdd(base + 3, (float)v[3]);
}

// One wave per 16-row strip; wave computes all NTILES 16x16 output tiles.
// A fragments (agg * norm_in, converted to bf16) are built ONCE, reused across col tiles.
// r = (agg * norm_in) @ W + b ; RELU_SCALE: store bf16(relu(r) * norm_out) for next layer,
// else store f32.
template <int NTILES, bool RELU_SCALE>
__global__ __launch_bounds__(32)
void gcn_gemm(const float* __restrict__ agg, const float* __restrict__ norm_in,
              const __bf16* __restrict__ Wp, const float* __restrict__ bias,
              const float* __restrict__ norm_out,
              __bf16* __restrict__ hs_out, float* __restrict__ f32_out) {
  const int lane = threadIdx.x;
  const int m    = lane & 15;
  const int half = lane >> 4;
  const int r0   = blockIdx.x * 16;

  const int rowA = r0 + m;
  const float nrm = norm_in[rowA];
  const float* arow = agg + (long)rowA * 128;

  // Build all four 16x32 bf16 A fragments once (ISA lane layout):
  //  a[kt][0..7]  = A[m][kt*32 + half*8 + 0..7]
  //  a[kt][8..15] = A[m][kt*32 + 16 + half*8 + 0..7]
  v16bf a[4];
#pragma unroll
  for (int kt = 0; kt < 4; ++kt) {
    const int k0 = kt * 32;
    v4f f0 = *(const v4f*)(arow + k0 + half * 8);
    v4f f1 = *(const v4f*)(arow + k0 + half * 8 + 4);
    v4f f2 = *(const v4f*)(arow + k0 + 16 + half * 8);
    v4f f3 = *(const v4f*)(arow + k0 + 16 + half * 8 + 4);
#pragma unroll
    for (int i = 0; i < 4; ++i) {
      a[kt][i]      = (__bf16)(f0[i] * nrm);
      a[kt][4 + i]  = (__bf16)(f1[i] * nrm);
      a[kt][8 + i]  = (__bf16)(f2[i] * nrm);
      a[kt][12 + i] = (__bf16)(f3[i] * nrm);
    }
  }

#pragma unroll
  for (int nt = 0; nt < NTILES; ++nt) {
    v8f acc = {0.f, 0.f, 0.f, 0.f, 0.f, 0.f, 0.f, 0.f};
#pragma unroll
    for (int kt = 0; kt < 4; ++kt) {
      // pre-packed B tile: one contiguous 32B load per lane
      v16bf b = *(const v16bf*)(Wp + ((long)(kt * NTILES + nt) * 32 + lane) * 16);
      acc = __builtin_amdgcn_wmma_f32_16x16x32_bf16(false, a[kt], false, b,
                                                    (short)0, acc, false, false);
    }
    const int col = nt * 16 + m;
    const float bv = bias[col];
#pragma unroll
    for (int j = 0; j < 8; ++j) {
      const int row = r0 + half * 8 + j;  // C/D layout: lanes 0-15 -> M=j, lanes 16-31 -> M=8+j
      float v = acc[j] + bv;
      if (RELU_SCALE) {
        v = fmaxf(v, 0.0f);
        hs_out[(long)row * 128 + col] = (__bf16)(v * norm_out[row]);
      } else {
        f32_out[(long)row * (NTILES * 16) + col] = v;
      }
    }
  }
}

// ---------------- launcher ----------------

extern "C" void kernel_launch(void* const* d_in, const int* in_sizes, int n_in,
                              void* d_out, int out_size, void* d_ws, size_t ws_size,
                              hipStream_t stream) {
  const float* feat = (const float*)d_in[0];
  const int*   src  = (const int*)d_in[1];
  const int*   dst  = (const int*)d_in[2];
  const float* W0   = (const float*)d_in[3];
  const float* b0   = (const float*)d_in[4];
  const float* W1   = (const float*)d_in[5];
  const float* b1   = (const float*)d_in[6];
  const float* W2   = (const float*)d_in[7];
  const float* b2   = (const float*)d_in[8];
  float* out = (float*)d_out;

  char* ws = (char*)d_ws;
  // 256B-aligned carve-up of workspace
  float*  norm_out = (float*)(ws + 0);            // 170000 f32
  float*  norm_in  = (float*)(ws + 680192);       // 170000 f32
  float*  agg      = (float*)(ws + 1360384);      // 170000*128 f32 (87 MB, L2-resident)
  __bf16* hs       = (__bf16*)(ws + 88400384);    // 170000*128 bf16 (pre-scaled activations)
  __bf16* Wp       = (__bf16*)(ws + 131920384);   // 128*128 bf16 packed weights

  // degrees -> norms
  gcn_zero_f32<<<(NN + 255) / 256, 256, 0, stream>>>(norm_out, NN);
  gcn_zero_f32<<<(NN + 255) / 256, 256, 0, stream>>>(norm_in, NN);
  gcn_degree<<<(NE + 255) / 256, 256, 0, stream>>>(src, dst, norm_out, norm_in);
  gcn_norm_finalize<<<(NN + 255) / 256, 256, 0, stream>>>(norm_out, NN);
  gcn_norm_finalize<<<(NN + 255) / 256, 256, 0, stream>>>(norm_in, NN);

  // hs = bf16(feat * norm_out)   (170000*128 / 4 / 256 = 21250 exact)
  gcn_scale_feat<<<21250, 256, 0, stream>>>(feat, norm_out, hs);

  const float* Ws[3] = {W0, W1, W2};
  const float* bs[3] = {b0, b1, b2};
  const int cols[3]  = {HID_D, HID_D, OUT_D};

  for (int l = 0; l < 3; ++l) {
    const int C = cols[l];
    gcn_pack_w<<<(128 * C + 255) / 256, 256, 0, stream>>>(Ws[l], Wp, C);
    // zero agg: 170000*128 floats / 4 / 256 = 21250 blocks exact
    gcn_zero_v4<<<21250, 256, 0, stream>>>(agg);
    // scatter-add over edges: 1.2M edges, 8 edges per 256-thread block
    gcn_edge_scatter<<<NE / 8, 256, 0, stream>>>(src, dst, hs, agg);
    // WMMA GEMM: one wave per 16-row strip, all col tiles in-wave
    if (l < 2) {
      gcn_gemm<HID_D / 16, true><<<NN / 16, 32, 0, stream>>>(
          agg, norm_in, Wp, bs[l], norm_out, hs, nullptr);
    } else {
      gcn_gemm<OUT_D / 16, false><<<NN / 16, 32, 0, stream>>>(
          agg, norm_in, Wp, bs[l], norm_out, nullptr, out);
    }
  }
}
